// CnnLstmCrf_22548578304855
// MI455X (gfx1250) — matrix operand order repaired and checked by reading, not compiled, not run on hardware
//
#include <hip/hip_runtime.h>
#include <hip/hip_bf16.h>

// ---------------- problem constants ----------------
constexpr int NB   = 128;   // batch
constexpr int NS   = 256;   // sequence length
constexpr int NL   = 10;    // chars per word
constexpr int NTAGS= 32;
constexpr int TSTART = 30;
constexpr int TEND   = 31;
constexpr int WD   = 100;   // word emb dim
constexpr int CD   = 30;    // char emb dim
constexpr int NF   = 30;    // conv filters
constexpr int HD   = 100;   // lstm hidden
constexpr int LSTM_IN = WD + NF;         // 130
constexpr int KCAT = LSTM_IN + HD;       // 230 (x|h)
constexpr int KP   = 256;                // padded K for lstm gemm (8 k-tiles)
constexpr int KT   = KP / 32;            // 8
constexpr int NGATE= 4 * HD;             // 400
constexpr int NT   = NGATE / 16;         // 25 n-tiles
constexpr int HCATP= 224;                // padded 2*HD for projection gemm (7 k-tiles)
constexpr int PKT  = HCATP / 32;         // 7
constexpr int PNT  = NTAGS / 16;         // 2
constexpr float NEGV = -10000.0f;

typedef __attribute__((ext_vector_type(16))) _Float16 v16h;
typedef __attribute__((ext_vector_type(8)))  float    v8f;
typedef int v4i32 __attribute__((vector_size(16)));   // 128-bit payload type for async-LDS builtin

// CDNA5 async global->LDS path (guarded so compile never breaks on arity drift)
#if defined(__gfx1250__) && __has_builtin(__builtin_amdgcn_global_load_async_to_lds_b128) && \
    __has_builtin(__builtin_amdgcn_s_wait_asynccnt)
#define HAVE_ASYNC_LDS 1
#else
#define HAVE_ASYNC_LDS 0
#endif

union FragU { v16h v; float4 q[2]; };

// A-fragment (16x32 f16) per documented CDNA5 layout:
// lanes 0-15: row=lane,    K = kt*32 + {0..7, 16..23}
// lanes16-31: row=lane-16, K = kt*32 + {8..15,24..31}
__device__ __forceinline__ v16h load_fragA(const _Float16* base, int ld_halves, int kt, int lane) {
    int row = lane & 15;
    int khi = (lane & 16) ? 8 : 0;
    const _Float16* p = base + (size_t)row * ld_halves + kt * 32 + khi;
    FragU u;
    u.q[0] = *reinterpret_cast<const float4*>(p);
    u.q[1] = *reinterpret_cast<const float4*>(p + 16);
    return u.v;
}

// B-fragment: pre-packed lane-major (32 lanes x 16 halves contiguous)
__device__ __forceinline__ v16h load_fragB(const _Float16* frag, int lane) {
    const _Float16* p = frag + lane * 16;
    FragU u;
    u.q[0] = *reinterpret_cast<const float4*>(p);
    u.q[1] = *reinterpret_cast<const float4*>(p + 8);
    return u.v;
}

__device__ __forceinline__ float sigf(float x) { return 1.0f / (1.0f + __expf(-x)); }

// ---------------- kernel 1: embeddings + char CNN + maxpool -> x (f16, (S,B,160)) ----------------
__global__ __launch_bounds__(64) void embed_conv_kernel(
    const float* __restrict__ word_emb, const float* __restrict__ char_emb,
    const float* __restrict__ conv_w,   const float* __restrict__ conv_b,
    const int* __restrict__ word_x,     const int* __restrict__ char_x,
    _Float16* __restrict__ xh)
{
    int tok = blockIdx.x;            // tok = b*NS + t
    int b = tok / NS, t = tok % NS;
    int tid = threadIdx.x;

    __shared__ __align__(16) float ce[14][CD];        // rows: positions -2..11 (zero padded)
    __shared__ __align__(16) float wsm[NF * 3 * CD];

    for (int i = tid; i < NF * 3 * CD; i += 64) wsm[i] = conv_w[i];
    for (int i = tid; i < 14 * CD; i += 64) (&ce[0][0])[i] = 0.0f;
    __syncthreads();
    for (int i = tid; i < NL * CD; i += 64) {
        int l = i / CD, d = i % CD;
        int cid = char_x[(size_t)tok * NL + l];
        ce[2 + l][d] = char_emb[(size_t)cid * CD + d];
    }
    __syncthreads();

    _Float16* xrow = xh + ((size_t)t * NB + b) * 160;

    if (tid < NF) {
        int f = tid;
        float m = -3.4e38f;
        for (int p = 0; p < 12; ++p) {
            float s = conv_b[f];
            #pragma unroll
            for (int k = 0; k < 3; ++k)
                for (int d = 0; d < CD; ++d)
                    s += ce[p + k][d] * wsm[(f * 3 + k) * CD + d];
            m = fmaxf(m, s);
        }
        xrow[WD + f] = (_Float16)m;   // char feature cols 100..129
    }
    int wid = word_x[tok];
    for (int k = tid; k < WD; k += 64)
        xrow[k] = (_Float16)word_emb[(size_t)wid * WD + k];
    for (int k = LSTM_IN + tid; k < 160; k += 64)
        xrow[k] = (_Float16)0.0f;     // pad cols 130..159
}

// ---------------- weight packing into WMMA B-fragment order ----------------
__global__ void pack_lstm_weights(const float* __restrict__ wih, const float* __restrict__ whh,
                                  _Float16* __restrict__ out)   // KT*NT fragments of 512 halves
{
    int idx = blockIdx.x * blockDim.x + threadIdx.x;
    int total = KT * NT * 32;
    if (idx >= total) return;
    int lane = idx & 31;
    int fn = idx >> 5;            // kt*NT + nt
    int nt = fn % NT, kt = fn / NT;
    _Float16* dst = out + (size_t)fn * 512 + lane * 16;
    int n = nt * 16 + (lane & 15);
    int khalf = (lane & 16) ? 8 : 0;
    #pragma unroll
    for (int j = 0; j < 16; ++j) {
        int k = kt * 32 + ((j < 8) ? (khalf + j) : (16 + khalf + (j - 8)));
        float v = 0.0f;
        if (k < LSTM_IN)      v = wih[(size_t)n * LSTM_IN + k];
        else if (k < KCAT)    v = whh[(size_t)n * HD + (k - LSTM_IN)];
        dst[j] = (_Float16)v;
    }
}

__global__ void pack_b_generic(const float* __restrict__ W, int Kdim, int ktiles, int ntiles,
                               _Float16* __restrict__ out)
{
    int idx = blockIdx.x * blockDim.x + threadIdx.x;
    int total = ktiles * ntiles * 32;
    if (idx >= total) return;
    int lane = idx & 31;
    int fn = idx >> 5;            // kt*ntiles + nt
    int nt = fn % ntiles, kt = fn / ntiles;
    _Float16* dst = out + (size_t)fn * 512 + lane * 16;
    int n = nt * 16 + (lane & 15);
    int khalf = (lane & 16) ? 8 : 0;
    #pragma unroll
    for (int j = 0; j < 16; ++j) {
        int k = kt * 32 + ((j < 8) ? (khalf + j) : (16 + khalf + (j - 8)));
        float v = (k < Kdim) ? W[(size_t)n * Kdim + k] : 0.0f;
        dst[j] = (_Float16)v;
    }
}

__global__ void zero_pad_hcat(_Float16* __restrict__ hcat)
{
    int idx = blockIdx.x * blockDim.x + threadIdx.x;
    int tok = idx / (HCATP - 2 * HD), j = idx % (HCATP - 2 * HD);
    if (tok < NB * NS) hcat[(size_t)tok * HCATP + 2 * HD + j] = (_Float16)0.0f;
}

// ---------------- kernel 2: persistent BiLSTM (WMMA recurrence) ----------------
// grid = 16 blocks: dir = bid&1, batch-group = bid>>1 (16 batch rows each)
__global__ __launch_bounds__(256) void lstm_kernel(
    const _Float16* __restrict__ xh,      // (NS, NB, 160) f16
    const _Float16* __restrict__ wpack,   // 2 * KT*NT*512 halves
    const float* __restrict__ bih_f, const float* __restrict__ bhh_f,
    const float* __restrict__ bih_b, const float* __restrict__ bhh_b,
    _Float16* __restrict__ hcat)          // (NB*NS, 224) f16
{
    int dir = blockIdx.x & 1;
    int b0  = (blockIdx.x >> 1) * 16;
    int tid = threadIdx.x;
    int wave = tid >> 5, lane = tid & 31;

    __shared__ __align__(16) _Float16 Atile[16][KP];   // 8 KB
    __shared__ __align__(16) float gbuf[16][NGATE];    // 25.6 KB
    __shared__ float hbuf[16][HD];
    __shared__ float cbuf[16][HD];
    __shared__ float bsum[NGATE];

    const float* bih = dir ? bih_b : bih_f;
    const float* bhh = dir ? bhh_b : bhh_f;
    const _Float16* wbase = wpack + (size_t)dir * KT * NT * 512;

    for (int i = tid; i < NGATE; i += 256) bsum[i] = bih[i] + bhh[i];
    for (int i = tid; i < 16 * HD; i += 256) { (&hbuf[0][0])[i] = 0.0f; (&cbuf[0][0])[i] = 0.0f; }
    for (int i = tid; i < 16 * KP; i += 256) (&Atile[0][0])[i] = (_Float16)0.0f;
    __syncthreads();

    for (int step = 0; step < NS; ++step) {
        int tt = dir ? (NS - 1 - step) : step;

        // ---- fill A = [x_t | h] ----
#if HAVE_ASYNC_LDS
        {
            // async b128: one 16B chunk per thread covers A cols 0..127 (16 rows x 16 chunks)
            int r = tid >> 4;
            int c = tid & 15;
            const _Float16* gsrc = xh + ((size_t)tt * NB + (b0 + r)) * 160 + c * 8;
            _Float16* ldst = &Atile[r][c * 8];
            __builtin_amdgcn_global_load_async_to_lds_b128(
                (__attribute__((address_space(1))) v4i32*)gsrc,
                (__attribute__((address_space(3))) v4i32*)ldst, 0, 0);
        }
        if (tid < 16) {  // cols 128..129 (one dword per row)
            int r = tid;
            *reinterpret_cast<unsigned int*>(&Atile[r][128]) =
                *reinterpret_cast<const unsigned int*>(xh + ((size_t)tt * NB + (b0 + r)) * 160 + 128);
        }
        for (int i = tid; i < 16 * HD; i += 256) {   // cols 130..229 from h
            int r = i / HD, k = i % HD;
            Atile[r][LSTM_IN + k] = (_Float16)hbuf[r][k];
        }
        __builtin_amdgcn_s_wait_asynccnt(0);
#else
        for (int i = tid; i < 16 * LSTM_IN; i += 256) {
            int r = i / LSTM_IN, k = i % LSTM_IN;
            Atile[r][k] = xh[((size_t)tt * NB + (b0 + r)) * 160 + k];
        }
        for (int i = tid; i < 16 * HD; i += 256) {
            int r = i / HD, k = i % HD;
            Atile[r][LSTM_IN + k] = (_Float16)hbuf[r][k];
        }
#endif
        __syncthreads();

        // ---- g = A x Wcat^T (16 x 400), 25 n-tiles over 8 waves ----
        for (int nt = wave; nt < NT; nt += 8) {
            v8f acc = {};
            #pragma unroll
            for (int kt = 0; kt < KT; ++kt) {
                v16h a  = load_fragA(&Atile[0][0], KP, kt, lane);
                v16h bm = load_fragB(wbase + ((size_t)kt * NT + nt) * 512, lane);
                acc = __builtin_amdgcn_wmma_f32_16x16x32_f16(
                          false, a, false, bm, (short)0, acc, false, false);
            }
            int col = nt * 16 + (lane & 15);
            int rbase = (lane >> 4) * 8;
            #pragma unroll
            for (int r = 0; r < 8; ++r) gbuf[rbase + r][col] = acc[r];
        }
        __syncthreads();

        // ---- gates + state update ----
        for (int i = tid; i < 16 * HD; i += 256) {
            int r = i / HD, j = i % HD;
            float ig = gbuf[r][j]          + bsum[j];
            float fg = gbuf[r][HD + j]     + bsum[HD + j];
            float gg = gbuf[r][2 * HD + j] + bsum[2 * HD + j];
            float og = gbuf[r][3 * HD + j] + bsum[3 * HD + j];
            float c  = sigf(fg) * cbuf[r][j] + sigf(ig) * tanhf(gg);
            float h  = sigf(og) * tanhf(c);
            cbuf[r][j] = c;
            hbuf[r][j] = h;
            hcat[((size_t)(b0 + r) * NS + tt) * HCATP + dir * HD + j] = (_Float16)h;
        }
        __syncthreads();
    }
}

// ---------------- kernel 3: emission = hcat @ proj_w^T + proj_b (WMMA) ----------------
__global__ __launch_bounds__(256) void emission_kernel(
    const _Float16* __restrict__ hcat, const _Float16* __restrict__ ppack,
    const float* __restrict__ proj_b, float* __restrict__ emission)
{
    int wave = threadIdx.x >> 5, lane = threadIdx.x & 31;
    int task = blockIdx.x * 8 + wave;           // (NB*NS/16)*PNT tasks = 4096
    int mt = task >> 1, nt = task & 1;
    const _Float16* Abase = hcat + (size_t)mt * 16 * HCATP;
    v8f acc = {};
    #pragma unroll
    for (int kt = 0; kt < PKT; ++kt) {
        v16h a  = load_fragA(Abase, HCATP, kt, lane);
        v16h bm = load_fragB(ppack + ((size_t)kt * PNT + nt) * 512, lane);
        acc = __builtin_amdgcn_wmma_f32_16x16x32_f16(
                  false, a, false, bm, (short)0, acc, false, false);
    }
    int col = nt * 16 + (lane & 15);
    float pb = proj_b[col];
    int rbase = (lane >> 4) * 8 + mt * 16;
    #pragma unroll
    for (int r = 0; r < 8; ++r)
        emission[(size_t)(rbase + r) * NTAGS + col] = acc[r] + pb;
}

// ---------------- CRF ----------------
__global__ __launch_bounds__(128) void crf_score_kernel(
    const float* __restrict__ emission, const float* __restrict__ trans,
    const int* __restrict__ y, const int* __restrict__ mask,
    float* __restrict__ score_y)
{
    int b = threadIdx.x;
    if (b >= NB) return;
    float s = 0.0f;
    int prev = TSTART;
    for (int t = 0; t < NS; ++t) {
        int cur = y[b * NS + t];
        float m = (float)mask[b * NS + t];
        s += (emission[((size_t)b * NS + t) * NTAGS + cur] + trans[prev * NTAGS + cur]) * m;
        prev = cur;
    }
    s += trans[y[b * NS + NS - 1] * NTAGS + TEND];
    score_y[b] = s;
}

__global__ __launch_bounds__(256) void crf_forward_kernel(
    const float* __restrict__ emission, const float* __restrict__ trans,
    const int* __restrict__ mask, float* __restrict__ total)
{
    __shared__ float tsh[NTAGS * NTAGS];
    __shared__ float la[8][NTAGS];
    int tid = threadIdx.x;
    int bl = tid >> 5;           // local batch row (0..7)
    int j  = tid & 31;           // tag
    int b  = blockIdx.x * 8 + bl;

    for (int i = tid; i < NTAGS * NTAGS; i += 256) tsh[i] = trans[i];
    la[bl][j] = (j == TSTART) ? 0.0f : NEGV;
    __syncthreads();

    for (int t = 0; t < NS; ++t) {
        float em = emission[((size_t)b * NS + t) * NTAGS + j];
        float mx = -3.4e38f;
        #pragma unroll
        for (int i = 0; i < NTAGS; ++i) mx = fmaxf(mx, la[bl][i] + tsh[i * NTAGS + j]);
        float sum = 0.0f;
        #pragma unroll
        for (int i = 0; i < NTAGS; ++i) sum += __expf(la[bl][i] + tsh[i * NTAGS + j] - mx);
        float la2 = mx + __logf(sum) + em;
        float m = (float)mask[b * NS + t];
        float nv = la2 * m + la[bl][j] * (1.0f - m);
        __syncthreads();
        la[bl][j] = nv;
        __syncthreads();
    }
    float v = la[bl][j] + tsh[j * NTAGS + TEND];
    __syncthreads();
    la[bl][j] = v;
    __syncthreads();
    if (j == 0) {
        float mx = -3.4e38f;
        for (int i = 0; i < NTAGS; ++i) mx = fmaxf(mx, la[bl][i]);
        float sum = 0.0f;
        for (int i = 0; i < NTAGS; ++i) sum += __expf(la[bl][i] - mx);
        total[b] = mx + __logf(sum);
    }
}

__global__ __launch_bounds__(128) void loss_kernel(
    const float* __restrict__ score_y, const float* __restrict__ total, float* __restrict__ out)
{
    __shared__ float red[128];
    int b = threadIdx.x;
    red[b] = score_y[b] - total[b];
    __syncthreads();
    for (int s = 64; s > 0; s >>= 1) {
        if (b < s) red[b] += red[b + s];
        __syncthreads();
    }
    if (b == 0) out[0] = -red[0] / (float)NB;
}

// ---------------- host launcher ----------------
extern "C" void kernel_launch(void* const* d_in, const int* in_sizes, int n_in,
                              void* d_out, int out_size, void* d_ws, size_t ws_size,
                              hipStream_t stream)
{
    const float* word_emb = (const float*)d_in[0];
    const float* char_emb = (const float*)d_in[1];
    const float* conv_w   = (const float*)d_in[2];
    const float* conv_b   = (const float*)d_in[3];
    const float* wih_f    = (const float*)d_in[4];
    const float* whh_f    = (const float*)d_in[5];
    const float* bih_f    = (const float*)d_in[6];
    const float* bhh_f    = (const float*)d_in[7];
    const float* wih_b    = (const float*)d_in[8];
    const float* whh_b    = (const float*)d_in[9];
    const float* bih_b    = (const float*)d_in[10];
    const float* bhh_b    = (const float*)d_in[11];
    const float* proj_w   = (const float*)d_in[12];
    const float* proj_b   = (const float*)d_in[13];
    const float* trans    = (const float*)d_in[14];
    const int*   word_x   = (const int*)d_in[15];
    const int*   char_x   = (const int*)d_in[16];
    const int*   y        = (const int*)d_in[17];
    const int*   mask     = (const int*)d_in[18];
    float* out = (float*)d_out;

    char* wsb = (char*)d_ws;
    size_t off = 0;
    auto take = [&](size_t bytes) -> char* {
        char* p = wsb + off;
        off += (bytes + 255) & ~(size_t)255;
        return p;
    };
    _Float16* xh     = (_Float16*)take((size_t)NS * NB * 160 * sizeof(_Float16));
    _Float16* wpack  = (_Float16*)take((size_t)2 * KT * NT * 512 * sizeof(_Float16));
    _Float16* ppack  = (_Float16*)take((size_t)PKT * PNT * 512 * sizeof(_Float16));
    _Float16* hcat   = (_Float16*)take((size_t)NB * NS * HCATP * sizeof(_Float16));
    float*    emis   = (float*)take((size_t)NB * NS * NTAGS * sizeof(float));
    float*    score  = (float*)take((size_t)NB * sizeof(float));
    float*    total  = (float*)take((size_t)NB * sizeof(float));

    // 1. embeddings + char CNN -> x (f16)
    embed_conv_kernel<<<NB * NS, 64, 0, stream>>>(word_emb, char_emb, conv_w, conv_b,
                                                  word_x, char_x, xh);
    // 2. weight packing
    {
        int th = KT * NT * 32;
        pack_lstm_weights<<<(th + 255) / 256, 256, 0, stream>>>(wih_f, whh_f, wpack);
        pack_lstm_weights<<<(th + 255) / 256, 256, 0, stream>>>(wih_b, whh_b,
                                                                wpack + (size_t)KT * NT * 512);
        int pt = PKT * PNT * 32;
        pack_b_generic<<<(pt + 63) / 64, 64, 0, stream>>>(proj_w, 2 * HD, PKT, PNT, ppack);
        int zp = NB * NS * (HCATP - 2 * HD);
        zero_pad_hcat<<<(zp + 255) / 256, 256, 0, stream>>>(hcat);
    }
    // 3. BiLSTM recurrence (persistent, WMMA)
    lstm_kernel<<<16, 256, 0, stream>>>(xh, wpack, bih_f, bhh_f, bih_b, bhh_b, hcat);
    // 4. emission projection (WMMA)
    emission_kernel<<<(NB * NS / 16) * PNT / 8, 256, 0, stream>>>(hcat, ppack, proj_b, emis);
    // 5. CRF
    crf_score_kernel<<<1, 128, 0, stream>>>(emis, trans, y, mask, score);
    crf_forward_kernel<<<NB / 8, 256, 0, stream>>>(emis, trans, mask, total);
    loss_kernel<<<1, 128, 0, stream>>>(score, total, out);

    (void)in_sizes; (void)n_in; (void)out_size; (void)ws_size;
}
